// Net_28561532518581
// MI455X (gfx1250) — matrix operand order repaired and checked by reading, not compiled, not run on hardware
//
#include <hip/hip_runtime.h>

typedef __attribute__((ext_vector_type(16))) _Float16 v16h;
typedef __attribute__((ext_vector_type(8)))  _Float16 v8h;
typedef __attribute__((ext_vector_type(8)))  float    v8f;

#define BETA    0.95f
#define THRESH  1.0f
#define T_STEPS 200
#define N_IN    80
#define K1P     96      // layer-1 K padded to multiple of 32
#define HID     256
#define NCLS    2
#define BATCH   4096
#define ROWS    16      // batch rows per workgroup
#define NTHREADS 256    // 8 waves

// LDS (halves): sW2[256*256] sW3[256*256] sW4[2*256] s1[16*256] s2 s3 xt[16*96]
#define SMEM_HALVES (2*65536 + 512 + 3*4096 + 1536)   // 145408 halves = 290816 B

__device__ inline v8f zero8() {
  v8f v;
#pragma unroll
  for (int i = 0; i < 8; ++i) v[i] = 0.0f;
  return v;
}

// Build the 16-half A operand for V_WMMA_F32_16X16X32_F16 from row-major [row][K]
// storage: lane holds row (lane&15), contiguous run [kk+16*hi .. +15].
__device__ inline v16h load_a(const _Float16* __restrict__ base, int rowStride,
                              int row, int kk, int hi) {
  const v8h a0 = *(const v8h*)(base + row * rowStride + kk + 16 * hi);
  const v8h a1 = *(const v8h*)(base + row * rowStride + kk + 16 * hi + 8);
  return __builtin_shufflevector(a0, a1, 0,1,2,3,4,5,6,7,8,9,10,11,12,13,14,15);
}

// Hidden layer (K=256): GEMM over 2 owned N-tiles + LIF update + spike emit.
__device__ inline void layer_hidden(const _Float16* __restrict__ sIn,
                                    const _Float16* __restrict__ Wlds,
                                    const float bias[2], v8f m[2],
                                    _Float16* __restrict__ sOut,
                                    int wave, int n15, int hi) {
#pragma unroll
  for (int nt = 0; nt < 2; ++nt) {
    const int cb = wave * 32 + nt * 16;
    v8f acc;
#pragma unroll
    for (int v = 0; v < 8; ++v) acc[v] = bias[nt];
#pragma unroll
    for (int kk = 0; kk < HID; kk += 32) {
      v16h A = load_a(sIn, HID, n15, kk, hi);
      v16h B = *(const v16h*)(Wlds + (cb + n15) * HID + kk + 16 * hi);
      acc = __builtin_amdgcn_wmma_f32_16x16x32_f16(false, A, false, B,
                                                   (short)0, acc, false, false);
    }
    v8f mm = m[nt];
#pragma unroll
    for (int v = 0; v < 8; ++v) {
      float mv = mm[v];
      const float reset = (mv > THRESH) ? THRESH : 0.0f;
      mv = BETA * mv + acc[v] - reset;
      mm[v] = mv;
      sOut[(v + 8 * hi) * HID + cb + n15] = (_Float16)((mv > THRESH) ? 1.0f : 0.0f);
    }
    m[nt] = mm;
  }
}

// One-time: W1 [256,80] f32 -> [256,96] f16 (zero-padded K).
__global__ void prep_w1(const float* __restrict__ W1, _Float16* __restrict__ W1h) {
  int idx = blockIdx.x * 256 + threadIdx.x;
  if (idx < 256 * K1P) {
    int n = idx / K1P, k = idx - n * K1P;
    W1h[idx] = (k < N_IN) ? (_Float16)W1[n * N_IN + k] : (_Float16)0.0f;
  }
}

__global__ __launch_bounds__(NTHREADS)
void snn_persistent(const float* __restrict__ x,
                    const _Float16* __restrict__ W1h,
                    const float* __restrict__ b1,
                    const float* __restrict__ W2, const float* __restrict__ b2,
                    const float* __restrict__ W3, const float* __restrict__ b3,
                    const float* __restrict__ W4, const float* __restrict__ b4,
                    float* __restrict__ out) {
  extern __shared__ _Float16 smem[];
  _Float16* sW2 = smem;              // 65536
  _Float16* sW3 = sW2 + 65536;       // 65536
  _Float16* sW4 = sW3 + 65536;       // 512
  _Float16* s1  = sW4 + 512;         // 4096
  _Float16* s2  = s1 + 4096;         // 4096
  _Float16* s3  = s2 + 4096;         // 4096
  _Float16* xt  = s3 + 4096;         // 1536 (16 x 96)

  const int tid  = threadIdx.x;
  const int lane = tid & 31;
  const int wave = tid >> 5;
  const int n15  = lane & 15;
  const int hi   = lane >> 4;
  const int row0 = blockIdx.x * ROWS;

  // ---- one-time LDS fill: weights f32 -> f16, zero x-tile pad columns ----
  for (int i = tid; i < 65536; i += NTHREADS) sW2[i] = (_Float16)W2[i];
  for (int i = tid; i < 65536; i += NTHREADS) sW3[i] = (_Float16)W3[i];
  for (int i = tid; i < NCLS * HID; i += NTHREADS) sW4[i] = (_Float16)W4[i];
  { int r = tid >> 4, c = N_IN + (tid & 15); xt[r * K1P + c] = (_Float16)0.0f; }
  __syncthreads();

  // ---- per-wave persistent state ----
  const int cb0 = wave * 32, cb1 = wave * 32 + 16;
  const float bias1[2] = { b1[cb0 + n15], b1[cb1 + n15] };
  const float bias2[2] = { b2[cb0 + n15], b2[cb1 + n15] };
  const float bias3[2] = { b3[cb0 + n15], b3[cb1 + n15] };
  const float bias4 = (n15 < NCLS) ? b4[n15] : 0.0f;

  v8f m1[2] = { zero8(), zero8() };
  v8f m2[2] = { zero8(), zero8() };
  v8f m3[2] = { zero8(), zero8() };
  v8f m4 = zero8();

  for (int t = 0; t < T_STEPS; ++t) {
    // ---- stage x_t tile (transpose strided x[b][i][t] -> f16 [16][96]) ----
    for (int idx = tid; idx < ROWS * N_IN; idx += NTHREADS) {
      int r = idx / N_IN, i = idx - r * N_IN;
      xt[r * K1P + i] =
          (_Float16)x[((size_t)(row0 + r) * N_IN + i) * T_STEPS + t];
    }
    __syncthreads();   // also fences last step's L4 read of s3 vs this step's L3

    // ---- layer 1: cur1 = x_t @ W1.T + b1 (K=96 padded) ----
#pragma unroll
    for (int nt = 0; nt < 2; ++nt) {
      const int cb = wave * 32 + nt * 16;
      v8f acc;
#pragma unroll
      for (int v = 0; v < 8; ++v) acc[v] = bias1[nt];
#pragma unroll
      for (int kk = 0; kk < K1P; kk += 32) {
        v16h A = load_a(xt, K1P, n15, kk, hi);
        v16h B = *(const v16h*)(W1h + (size_t)(cb + n15) * K1P + kk + 16 * hi);
        acc = __builtin_amdgcn_wmma_f32_16x16x32_f16(false, A, false, B,
                                                     (short)0, acc, false, false);
      }
      v8f mm = m1[nt];
#pragma unroll
      for (int v = 0; v < 8; ++v) {
        float mv = mm[v];
        const float reset = (mv > THRESH) ? THRESH : 0.0f;
        mv = BETA * mv + acc[v] - reset;
        mm[v] = mv;
        s1[(v + 8 * hi) * HID + cb + n15] = (_Float16)((mv > THRESH) ? 1.0f : 0.0f);
      }
      m1[nt] = mm;
    }
    __syncthreads();

    // ---- layer 2 / layer 3 ----
    layer_hidden(s1, sW2, bias2, m2, s2, wave, n15, hi);
    __syncthreads();
    layer_hidden(s2, sW3, bias3, m3, s3, wave, n15, hi);
    __syncthreads();

    // ---- layer 4 (wave 0 only): N=2 padded to one 16-wide tile ----
    if (wave == 0) {
      v8f acc;
#pragma unroll
      for (int v = 0; v < 8; ++v) acc[v] = bias4;
#pragma unroll
      for (int kk = 0; kk < HID; kk += 32) {
        v16h A = load_a(s3, HID, n15, kk, hi);
        v16h B;
        if (n15 < NCLS) {
          B = *(const v16h*)(sW4 + n15 * HID + kk + 16 * hi);
        } else {
#pragma unroll
          for (int i = 0; i < 16; ++i) B[i] = (_Float16)0.0f;
        }
        acc = __builtin_amdgcn_wmma_f32_16x16x32_f16(false, A, false, B,
                                                     (short)0, acc, false, false);
      }
#pragma unroll
      for (int v = 0; v < 8; ++v) {
        float mv = m4[v];
        const float reset = (mv > THRESH) ? THRESH : 0.0f;
        mv = BETA * mv + acc[v] - reset;
        m4[v] = mv;
        if (n15 < NCLS) {
          const int row = v + 8 * hi;
          out[(size_t)t * BATCH * NCLS + (size_t)(row0 + row) * NCLS + n15] =
              (mv > THRESH) ? 1.0f : 0.0f;
        }
      }
    }
    // next iteration's staging barrier orders s3 reuse
  }
}

extern "C" void kernel_launch(void* const* d_in, const int* in_sizes, int n_in,
                              void* d_out, int out_size, void* d_ws, size_t ws_size,
                              hipStream_t stream) {
  const float* x  = (const float*)d_in[0];
  const float* W1 = (const float*)d_in[1];
  const float* b1 = (const float*)d_in[2];
  const float* W2 = (const float*)d_in[3];
  const float* b2 = (const float*)d_in[4];
  const float* W3 = (const float*)d_in[5];
  const float* b3 = (const float*)d_in[6];
  const float* W4 = (const float*)d_in[7];
  const float* b4 = (const float*)d_in[8];
  float* out = (float*)d_out;

  _Float16* W1h = (_Float16*)d_ws;   // 256*96*2 = 48 KB scratch

  prep_w1<<<(256 * K1P + 255) / 256, 256, 0, stream>>>(W1, W1h);

  const size_t smemBytes = SMEM_HALVES * sizeof(_Float16);  // 290816 B < 320 KB
  snn_persistent<<<BATCH / ROWS, NTHREADS, smemBytes, stream>>>(
      x, W1h, b1, W2, b2, W3, b3, W4, b4, out);
}
